// LSTMClassification_15083925144233
// MI455X (gfx1250) — compile-verified, matrix operand order checked by reading, not run on hardware
//
#include <hip/hip_runtime.h>

typedef __attribute__((ext_vector_type(16))) _Float16 v16h;
typedef __attribute__((ext_vector_type(8)))  _Float16 v8h;
typedef __attribute__((ext_vector_type(8)))  float    v8f;

#define B_   64
#define T_   512
#define IN_  128
#define H_   512
#define NG   (4 * H_)      // 2048 gate columns
#define NWG  32            // persistent workgroups
#define UPW  (H_ / NWG)    // 16 hidden units per workgroup
#define LDSPAD 8           // halfs of per-row padding (16B) -> kills LDS bank conflicts

// ---------------------------------------------------------------------------
// Conversion / prep kernels (off the critical path, fully parallel)
// ---------------------------------------------------------------------------
__global__ void cvt_f32_to_f16(const float* __restrict__ in,
                               _Float16* __restrict__ out, int n) {
  int i = blockIdx.x * blockDim.x + threadIdx.x;
  if (i < n) out[i] = (_Float16)in[i];
}

// x [B,T,IN] f32 -> x16 [T,B,IN] f16 (time-major so each step's A rows are contiguous)
__global__ void cvt_x_tmajor(const float* __restrict__ x,
                             _Float16* __restrict__ xo) {
  int i = blockIdx.x * blockDim.x + threadIdx.x;
  int n = T_ * B_ * IN_;
  if (i >= n) return;
  int c = i % IN_;
  int b = (i / IN_) % B_;
  int t = i / (IN_ * B_);
  xo[i] = (_Float16)x[((size_t)b * T_ + t) * IN_ + c];
}

__global__ void add_bias(const float* __restrict__ a, const float* __restrict__ b,
                         float* __restrict__ o, int n) {
  int i = blockIdx.x * blockDim.x + threadIdx.x;
  if (i < n) o[i] = a[i] + b[i];
}

// ---------------------------------------------------------------------------
// Fragment load — CDNA5 16-bit A/B WMMA layout (ISA 7.12.2):
// lane L (L%16 = row), halfs: e in [0,8)  -> K = kb + 8*(L/16) + e
//                      e in [8,16) -> K = kb + 16 + 8*(L/16) + (e-8)
// => two contiguous 16-byte loads per lane from a row-major [row, K] source.
// Works for global (A operand) and LDS (staged weights; clang infers
// addrspace(3) after inlining and emits ds_load_b128).
// ---------------------------------------------------------------------------
union FragU { v16h v; struct { v8h lo, hi; } p; };

__device__ __forceinline__ v16h load_frag(const _Float16* __restrict__ row,
                                          int kb, int lhalf) {
  FragU u;
  u.p.lo = *(const v8h*)(row + kb + 8 * lhalf);
  u.p.hi = *(const v8h*)(row + kb + 16 + 8 * lhalf);
  return u.v;
}

__device__ __forceinline__ float sigm(float x) { return 1.0f / (1.0f + __expf(-x)); }

// ---------------------------------------------------------------------------
// Persistent LSTM layer kernel.
//  grid = NWG (32) workgroups x 256 threads (8 wave32).
//  WG w owns hidden units [16w, 16w+16) and computes gate columns
//  {g*512 + 16w .. +16} for g = 0..3, so the c/h update is WG-local.
//  The WG's 64 weight rows (W_ih | W_hh concatenated per gate-column) are
//  staged ONCE into LDS (83/129 KB) and reused for all 512 timesteps:
//  the serial recurrent loop then feeds WMMA B-operands at LDS latency
//  with zero per-step L2 weight traffic.
// ---------------------------------------------------------------------------
__global__ void __launch_bounds__(256)
lstm_persistent(const _Float16* __restrict__ xseq,  // [T, B, INK] f16
                int INK,
                const _Float16* __restrict__ Wih,   // [4H, INK] f16 row-major
                const _Float16* __restrict__ Whh,   // [4H, H]   f16 row-major
                const float*    __restrict__ bias,  // [4H] (b_ih + b_hh)
                _Float16*       __restrict__ hbuf,  // [2, B, H] f16 (double buffer, pre-zeroed)
                _Float16*       __restrict__ seq_out, // null or [T, B, H] f16
                float*          __restrict__ last_out,// null or [B, H] f32
                int* __restrict__ sync_cnt,
                int* __restrict__ sync_gen) {
  extern __shared__ _Float16 w_lds[];   // [64][INK + H + LDSPAD] staged weights
  __shared__ float lds_g[4][B_][UPW];   // gate preactivations, 16 KB
  __shared__ float lds_c[B_][UPW];      // cell state (WG-local for whole layer), 4 KB
  __shared__ float lds_b[4][UPW];       // bias slice

  const int tid   = threadIdx.x;
  const int lane  = tid & 31;
  const int wv    = tid >> 5;         // wave 0..7
  const int lhalf = lane >> 4;        // 0|1
  const int lrow  = lane & 15;        // M for A / N for B / N for C
  const int m0    = (wv & 3) * 16;    // batch tile base (4 tiles cover B=64)
  const int gp    = wv >> 2;          // gate pair: 0 -> (i,f), 1 -> (g,o)
  const int gA    = gp * 2;
  const int gB    = gp * 2 + 1;
  const int hu0   = blockIdx.x * UPW; // first hidden unit owned by this WG
  const int LROW  = INK + H_ + LDSPAD; // padded LDS row stride (halfs)

  // ---- one-time weight stage into LDS: 64 rows, wave wv copies rows wv+8k
  for (int lc = wv; lc < 4 * UPW; lc += 8) {
    int g = lc >> 4, u = lc & 15;
    const _Float16* gih = Wih + (size_t)(g * H_ + hu0 + u) * INK;
    const _Float16* ghh = Whh + (size_t)(g * H_ + hu0 + u) * H_;
    _Float16* dst = w_lds + (size_t)lc * LROW;
    for (int c = lane; c < INK / 8; c += 32)
      ((v8h*)dst)[c] = ((const v8h*)gih)[c];
    _Float16* dsth = dst + INK;
    for (int c = lane; c < H_ / 8; c += 32)
      ((v8h*)dsth)[c] = ((const v8h*)ghh)[c];
  }
  for (int i = tid; i < B_ * UPW; i += blockDim.x) lds_c[i / UPW][i % UPW] = 0.0f;
  for (int i = tid; i < 4 * UPW; i += blockDim.x)
    lds_b[i / UPW][i % UPW] = bias[(i / UPW) * H_ + hu0 + (i % UPW)];
  __syncthreads();

  // Per-wave B-operand rows in LDS (lane's gate-column), constant over time.
  const _Float16* BrowA = w_lds + (size_t)(gA * UPW + lrow) * LROW;
  const _Float16* BrowB = w_lds + (size_t)(gB * UPW + lrow) * LROW;

  for (int t = 0; t < T_; ++t) {
    const _Float16* xt = xseq + (size_t)t * B_ * INK;
    const _Float16* hp = hbuf + (size_t)(t & 1) * B_ * H_;
    _Float16*       hn = hbuf + (size_t)((t + 1) & 1) * B_ * H_;

    // Warm next step's x rows in WGP$/L2 while this step computes + barriers.
    if (t + 1 < T_) {
      const _Float16* nxrow = xseq + (size_t)(t + 1) * B_ * INK
                                   + (size_t)(m0 + lrow) * INK;
      for (int pc = lhalf; pc * 64 < INK * 2; pc += 2)
        __builtin_prefetch(nxrow + pc * 32, 0, 1);   // 64B granules
    }

    v8f accA = {};
    v8f accB = {};

    // Input contribution: A = x_t rows [m0..m0+16), B from LDS rows [0, INK)
    {
      const _Float16* arow = xt + (size_t)(m0 + lrow) * INK;
      for (int kb = 0; kb < INK; kb += 32) {
        v16h a  = load_frag(arow, kb, lhalf);
        v16h bA = load_frag(BrowA, kb, lhalf);
        v16h bB = load_frag(BrowB, kb, lhalf);
        accA = __builtin_amdgcn_wmma_f32_16x16x32_f16(false, a, false, bA,
                                                      (short)0, accA, false, false);
        accB = __builtin_amdgcn_wmma_f32_16x16x32_f16(false, a, false, bB,
                                                      (short)0, accB, false, false);
      }
    }
    // Recurrent contribution: A = h_{t-1} rows, B from LDS rows [INK, INK+H)
    {
      const _Float16* arow  = hp + (size_t)(m0 + lrow) * H_;
      const _Float16* BrAh  = BrowA + INK;
      const _Float16* BrBh  = BrowB + INK;
      for (int kb = 0; kb < H_; kb += 32) {
        v16h a  = load_frag(arow, kb, lhalf);
        v16h bA = load_frag(BrAh, kb, lhalf);
        v16h bB = load_frag(BrBh, kb, lhalf);
        accA = __builtin_amdgcn_wmma_f32_16x16x32_f16(false, a, false, bA,
                                                      (short)0, accA, false, false);
        accB = __builtin_amdgcn_wmma_f32_16x16x32_f16(false, a, false, bB,
                                                      (short)0, accB, false, false);
      }
    }

    // Scatter C tiles to LDS: VGPR r -> batch row m0 + r + 8*lhalf, col = lrow.
#pragma unroll
    for (int r = 0; r < 8; ++r) {
      int brow = m0 + r + 8 * lhalf;
      lds_g[gA][brow][lrow] = accA[r];
      lds_g[gB][brow][lrow] = accB[r];
    }
    __syncthreads();

    // Pointwise LSTM cell for this WG's 64x16 (batch x unit) slab.
    for (int i = tid; i < B_ * UPW; i += blockDim.x) {
      int b = i / UPW, u = i % UPW;
      float ig = sigm(lds_g[0][b][u] + lds_b[0][u]);
      float fg = sigm(lds_g[1][b][u] + lds_b[1][u]);
      float gg = tanhf(lds_g[2][b][u] + lds_b[2][u]);
      float og = sigm(lds_g[3][b][u] + lds_b[3][u]);
      float c  = fg * lds_c[b][u] + ig * gg;
      lds_c[b][u] = c;
      float h  = og * tanhf(c);
      _Float16 h16 = (_Float16)h;
      hn[(size_t)b * H_ + hu0 + u] = h16;
      if (seq_out)  seq_out[((size_t)t * B_ + b) * H_ + hu0 + u] = h16;
      if (last_out && t == T_ - 1) last_out[(size_t)b * H_ + hu0 + u] = h;
    }

    // ---- grid-wide barrier (all 32 WGs co-resident) ----
    __threadfence();     // make hn writes device-visible
    __syncthreads();     // also protects lds_g reuse next step
    if (tid == 0) {
      int target = t + 1;
      int prev = __hip_atomic_fetch_add(sync_cnt, 1, __ATOMIC_ACQ_REL,
                                        __HIP_MEMORY_SCOPE_AGENT);
      if (prev == NWG - 1) {
        __hip_atomic_store(sync_cnt, 0, __ATOMIC_RELAXED, __HIP_MEMORY_SCOPE_AGENT);
        __hip_atomic_store(sync_gen, target, __ATOMIC_RELEASE, __HIP_MEMORY_SCOPE_AGENT);
      } else {
        while (__hip_atomic_load(sync_gen, __ATOMIC_ACQUIRE,
                                 __HIP_MEMORY_SCOPE_AGENT) < target) {
          __builtin_amdgcn_s_sleep(1);
        }
      }
    }
    __syncthreads();
    __threadfence();     // acquire other WGs' h writes before next step's loads
  }
}

// ---------------------------------------------------------------------------
// Final FC head: logits[b] = h_last[b,:] . W_fc + b_fc   (tiny; 64x512 dot)
// ---------------------------------------------------------------------------
__global__ void fc_kernel(const float* __restrict__ hl, const float* __restrict__ Wfc,
                          const float* __restrict__ bfc, float* __restrict__ out) {
  int b = blockIdx.x * blockDim.x + threadIdx.x;
  if (b >= B_) return;
  float s = 0.0f;
  for (int k = 0; k < H_; ++k) s += hl[(size_t)b * H_ + k] * Wfc[k];
  out[b] = s + bfc[0];
}

// ---------------------------------------------------------------------------
// Host launcher
// ---------------------------------------------------------------------------
extern "C" void kernel_launch(void* const* d_in, const int* in_sizes, int n_in,
                              void* d_out, int out_size, void* d_ws, size_t ws_size,
                              hipStream_t stream) {
  const float* x    = (const float*)d_in[0];
  const float* Wih0 = (const float*)d_in[1];
  const float* Whh0 = (const float*)d_in[2];
  const float* bih0 = (const float*)d_in[3];
  const float* bhh0 = (const float*)d_in[4];
  const float* Wih1 = (const float*)d_in[5];
  const float* Whh1 = (const float*)d_in[6];
  const float* bih1 = (const float*)d_in[7];
  const float* bhh1 = (const float*)d_in[8];
  const float* Wfc  = (const float*)d_in[9];
  const float* bfc  = (const float*)d_in[10];

  // Workspace carve (~49 MB total), 256-byte aligned chunks.
  char* ws = (char*)d_ws;
  size_t off = 0;
  auto carve = [&](size_t bytes) -> char* {
    char* p = ws + off;
    off += (bytes + 255) & ~(size_t)255;
    return p;
  };
  _Float16* wih0h = (_Float16*)carve((size_t)NG * IN_ * 2);
  _Float16* whh0h = (_Float16*)carve((size_t)NG * H_ * 2);
  _Float16* wih1h = (_Float16*)carve((size_t)NG * H_ * 2);
  _Float16* whh1h = (_Float16*)carve((size_t)NG * H_ * 2);
  float*    bias0 = (float*)carve((size_t)NG * 4);
  float*    bias1 = (float*)carve((size_t)NG * 4);
  _Float16* x16   = (_Float16*)carve((size_t)T_ * B_ * IN_ * 2);
  _Float16* h0seq = (_Float16*)carve((size_t)T_ * B_ * H_ * 2);
  _Float16* hbufA = (_Float16*)carve((size_t)2 * B_ * H_ * 2);
  _Float16* hbufB = (_Float16*)carve((size_t)2 * B_ * H_ * 2);
  float*    hlast = (float*)carve((size_t)B_ * H_ * 4);
  int*      ctrl  = (int*)carve(4 * sizeof(int)); // {cnt0, gen0, cnt1, gen1}

  // Weight / input conversion (parallel, off critical path).
  int n;
  n = NG * IN_;
  cvt_f32_to_f16<<<(n + 255) / 256, 256, 0, stream>>>(Wih0, wih0h, n);
  n = NG * H_;
  cvt_f32_to_f16<<<(n + 255) / 256, 256, 0, stream>>>(Whh0, whh0h, n);
  cvt_f32_to_f16<<<(n + 255) / 256, 256, 0, stream>>>(Wih1, wih1h, n);
  cvt_f32_to_f16<<<(n + 255) / 256, 256, 0, stream>>>(Whh1, whh1h, n);
  n = NG;
  add_bias<<<(n + 255) / 256, 256, 0, stream>>>(bih0, bhh0, bias0, n);
  add_bias<<<(n + 255) / 256, 256, 0, stream>>>(bih1, bhh1, bias1, n);
  n = T_ * B_ * IN_;
  cvt_x_tmajor<<<(n + 255) / 256, 256, 0, stream>>>(x, x16);

  // Zero h double-buffers and sync state every call (deterministic, capturable).
  hipMemsetAsync(hbufA, 0, (size_t)2 * B_ * H_ * 2, stream);
  hipMemsetAsync(hbufB, 0, (size_t)2 * B_ * H_ * 2, stream);
  hipMemsetAsync(ctrl, 0, 4 * sizeof(int), stream);

  // Dynamic LDS for staged weights: 64 padded rows of (INK + H + LDSPAD) halfs.
  size_t lds0 = (size_t)4 * UPW * (IN_ + H_ + LDSPAD) * sizeof(_Float16); //  ~83 KB
  size_t lds1 = (size_t)4 * UPW * (H_  + H_ + LDSPAD) * sizeof(_Float16); // ~129 KB
  (void)hipFuncSetAttribute((const void*)lstm_persistent,
                            hipFuncAttributeMaxDynamicSharedMemorySize, (int)lds1);

  // Layer 0: persistent, writes full h-sequence (f16) for layer 1.
  lstm_persistent<<<NWG, 256, lds0, stream>>>(x16, IN_, wih0h, whh0h, bias0,
                                              hbufA, h0seq, nullptr,
                                              ctrl + 0, ctrl + 1);
  // Layer 1: persistent, consumes h0seq, writes only the final h (f32).
  lstm_persistent<<<NWG, 256, lds1, stream>>>(h0seq, H_, wih1h, whh1h, bias1,
                                              hbufB, nullptr, hlast,
                                              ctrl + 2, ctrl + 3);
  // FC head.
  fc_kernel<<<1, 64, 0, stream>>>(hlast, Wfc, bfc, (float*)d_out);
}
